// MultiHeadedAttentionNeighbor_83159156785323
// MI455X (gfx1250) — matrix operand order repaired and checked by reading, not compiled, not run on hardware
//
#include <hip/hip_runtime.h>
#include <hip/hip_bf16.h>
#include <math.h>

// ---------------------------------------------------------------------------
// Problem constants (from reference)
// ---------------------------------------------------------------------------
#define BATCH 16
#define HEADS 8
#define SEQ   512
#define DMODEL 512
#define DK    64
#define NTOK  (BATCH * SEQ)      // 8192 tokens
#define BH    (BATCH * HEADS)    // 128 head-batches

typedef _Float16 half_t;
typedef __attribute__((ext_vector_type(16))) _Float16 v16h;
typedef __attribute__((ext_vector_type(8)))  _Float16 v8h;
typedef __attribute__((ext_vector_type(8)))  float    v8f;

// ---------------------------------------------------------------------------
// WMMA helper: D = A(16x32 f16) * B(32x16 f16) + C(16x16 f32)
// ---------------------------------------------------------------------------
static __device__ __forceinline__ v8f wmma16x16x32(v16h a, v16h b, v8f c) {
    return __builtin_amdgcn_wmma_f32_16x16x32_f16(
        /*neg_a=*/false, a, /*neg_b=*/false, b,
        /*c_mod=*/(short)0, c, /*reuse_a=*/false, /*reuse_b=*/false);
}

// A-matrix fragment (16x32, f16). Lane m<16 holds row m with
// K = kb..kb+15, kb = (lane>=16)*16.  Two 16B chunks.
static __device__ __forceinline__ v16h load_afrag_h(const half_t* base, int ldh,
                                                    int k0, int lane) {
    const int row = lane & 15;
    const int kb  = (lane >> 4) << 4;
    const half_t* p = base + (size_t)row * ldh + k0 + kb;
    v8h lo = *(const v8h*)p;
    v8h hi = *(const v8h*)(p + 8);
    v16h r;
#pragma unroll
    for (int j = 0; j < 8; ++j) { r[j] = lo[j]; r[j + 8] = hi[j]; }
    return r;
}

// B-matrix fragment (32x16, f16).  Lane n<16 holds column n, K = 0..15
// packed 2-per-VGPR; lanes 16..31 hold K = 16..31.  Source is "B^T"
// row-major: row n of `base` supplies column n, contiguous along K.
static __device__ __forceinline__ v16h load_bfrag_h(const half_t* base, int ldh,
                                                    int k0, int lane) {
    const int col = lane & 15;
    const int ks  = (lane >> 4) << 4;
    const half_t* p = base + (size_t)col * ldh + k0 + ks;
    v8h lo = *(const v8h*)p;
    v8h hi = *(const v8h*)(p + 8);
    v16h r;
#pragma unroll
    for (int j = 0; j < 8; ++j) { r[j] = lo[j]; r[j + 8] = hi[j]; }
    return r;
}

// ---------------------------------------------------------------------------
// Kernel 0: one-shot f32 -> f16 conversion (8 elements / thread, b128 I/O)
// ---------------------------------------------------------------------------
__global__ void __launch_bounds__(256)
cvt_f16_kernel(const float* __restrict__ src, half_t* __restrict__ dst, int n8) {
    const int i = blockIdx.x * 256 + threadIdx.x;
    if (i >= n8) return;
    const size_t base = (size_t)i * 8;
    const float4 a = ((const float4*)(src + base))[0];
    const float4 b = ((const float4*)(src + base))[1];
    v8h o;
    o[0] = (half_t)a.x; o[1] = (half_t)a.y; o[2] = (half_t)a.z; o[3] = (half_t)a.w;
    o[4] = (half_t)b.x; o[5] = (half_t)b.y; o[6] = (half_t)b.z; o[7] = (half_t)b.w;
    *(v8h*)(dst + base) = o;
}

// ---------------------------------------------------------------------------
// Kernel 1: projection GEMM (all-f16 operands, f32 accumulate).
// C[M=8192,N=512] = X[M,512] * W^T + bias, emitted f16 either head-major
// [BH][L][DK] (Q,K) or transposed [BH][DK][L] (V -> contiguous B-frags later).
// Block 256 thr = 8 waves; wave = 32 rows x 64 cols (2x4 WMMA tiles).
// ---------------------------------------------------------------------------
__global__ void __launch_bounds__(256)
proj_gemm_kernel(const half_t* __restrict__ X, const half_t* __restrict__ W,
                 const float* __restrict__ bias, half_t* __restrict__ dst,
                 int transposedV) {
    const int lane = threadIdx.x & 31;
    const int wave = threadIdx.x >> 5;
    const int m0 = blockIdx.y * 256 + wave * 32;
    const int n0 = blockIdx.x * 64;

    v8f acc[2][4];
#pragma unroll
    for (int r = 0; r < 2; ++r)
#pragma unroll
        for (int t = 0; t < 4; ++t) acc[r][t] = v8f{};

    const half_t* xrow0 = X + (size_t)m0 * DMODEL;
    const half_t* xrow1 = X + (size_t)(m0 + 16) * DMODEL;

    for (int k0 = 0; k0 < DMODEL; k0 += 32) {
        v16h a0 = load_afrag_h(xrow0, DMODEL, k0, lane);
        v16h a1 = load_afrag_h(xrow1, DMODEL, k0, lane);
#pragma unroll
        for (int t = 0; t < 4; ++t) {
            v16h b = load_bfrag_h(W + (size_t)(n0 + t * 16) * DMODEL, DMODEL, k0, lane);
            acc[0][t] = wmma16x16x32(a0, b, acc[0][t]);
            acc[1][t] = wmma16x16x32(a1, b, acc[1][t]);
        }
    }

    const int mb = (lane >> 4) << 3;
#pragma unroll
    for (int r = 0; r < 2; ++r) {
#pragma unroll
        for (int t = 0; t < 4; ++t) {
            const int n = n0 + t * 16 + (lane & 15);
            const int h = n >> 6, d = n & 63;
            const float bvv = bias[n];
#pragma unroll
            for (int j = 0; j < 8; ++j) {
                const int m = m0 + r * 16 + mb + j;
                const int bidx = m >> 9, l = m & 511;
                const half_t v = (half_t)(acc[r][t][j] + bvv);
                if (!transposedV)
                    dst[(((size_t)bidx * HEADS + h) * SEQ + l) * DK + d] = v;
                else
                    dst[(((size_t)bidx * HEADS + h) * DK + d) * SEQ + l] = v;
            }
        }
    }
}

// ---------------------------------------------------------------------------
// Kernel 2: per-row gammas:  g = q @ Wg^T + bg (DK->2), qp = (|g|+1)^-2.
// ---------------------------------------------------------------------------
__global__ void __launch_bounds__(256)
gamma_kernel(const half_t* __restrict__ Qh, const float* __restrict__ Wg,
             const float* __restrict__ bg, float* __restrict__ QP) {
    const int gi = blockIdx.x * 256 + threadIdx.x;   // 0 .. BH*SEQ-1
    const half_t* q = Qh + (size_t)gi * DK;
    float g0 = bg[0], g1 = bg[1];
#pragma unroll
    for (int c = 0; c < DK / 8; ++c) {
        v8h qv = *(const v8h*)(q + c * 8);
#pragma unroll
        for (int u = 0; u < 8; ++u) {
            const float f = (float)qv[u];
            g0 += f * Wg[c * 8 + u];
            g1 += f * Wg[DK + c * 8 + u];
        }
    }
    const float t0 = 1.f + fabsf(g0);
    const float t1 = 1.f + fabsf(g1);
    QP[(size_t)gi * 2 + 0] = 1.f / (t0 * t0);
    QP[(size_t)gi * 2 + 1] = 1.f / (t1 * t1);
}

// ---------------------------------------------------------------------------
// Kernel 3: fused attention, one block per (bh, 32-row strip).
// LDS: sS 32x520 f32 (scores, padded pitch), sP 32x528 f16 (normalized attn).
// Phase A: S = Q K^T via WMMA           (8 waves x 2x2 WMMA x 8 col groups)
// Phase B: rowmax -> exp(s-max), adjacency mix, row-sum normalize (wave32
//          __shfl_xor reductions; attn/max(softmax) == exp(s - rowmax))
// Phase C: O = A V via WMMA (V pre-transposed so B-frags are contiguous)
// ---------------------------------------------------------------------------
#define SPITCH 520
#define PPITCH 528
#define SMEM_S_BYTES (32 * SPITCH * 4)
#define SMEM_TOTAL   (SMEM_S_BYTES + 32 * PPITCH * 2)

__global__ void __launch_bounds__(256)
attn_kernel(const half_t* __restrict__ Qh, const half_t* __restrict__ Kh,
            const half_t* __restrict__ Vt, const float* __restrict__ QP,
            const float* __restrict__ lam1, half_t* __restrict__ Octx) {
    extern __shared__ char smem[];
    float*  sS = (float*)smem;
    half_t* sP = (half_t*)(smem + SMEM_S_BYTES);

    const int lane = threadIdx.x & 31;
    const int wave = threadIdx.x >> 5;
    const int bh   = blockIdx.y;
    const int row0 = blockIdx.x * 32;
    const int bb = bh >> 3, hh = bh & 7;

    const half_t* Qbh = Qh + (size_t)bh * SEQ * DK;
    const half_t* Kbh = Kh + (size_t)bh * SEQ * DK;
    const half_t* Vbh = Vt + (size_t)bh * DK * SEQ;

    // Warm the V panel (64KB) into cache while phases A/B execute
    // (global_prefetch_b8; one cacheline per thread covers half the panel).
    __builtin_prefetch(Vbh + (size_t)threadIdx.x * 128, 0, 0);

    // ---- Phase A: scores -> LDS --------------------------------------
    {
        const int rsub = wave & 1;          // which 16-row half of the strip
        const int cgrp = wave >> 1;         // 4 column groups per row half
        const half_t* qbase = Qbh + (size_t)(row0 + rsub * 16) * DK;
        v16h a0 = load_afrag_h(qbase, DK, 0,  lane);
        v16h a1 = load_afrag_h(qbase, DK, 32, lane);
        const int nloc = lane & 15;
        const int mb   = (lane >> 4) << 3;
#pragma unroll
        for (int t = 0; t < 8; ++t) {
            const int ct = cgrp + 4 * t;    // col tile 0..31
            const half_t* kbase = Kbh + (size_t)(ct * 16) * DK;
            v16h b0 = load_bfrag_h(kbase, DK, 0,  lane);
            v16h b1 = load_bfrag_h(kbase, DK, 32, lane);
            v8f acc = {};
            acc = wmma16x16x32(a0, b0, acc);
            acc = wmma16x16x32(a1, b1, acc);
            const int n = ct * 16 + nloc;
#pragma unroll
            for (int j = 0; j < 8; ++j)
                sS[(size_t)(rsub * 16 + mb + j) * SPITCH + n] = acc[j];
        }
    }
    __syncthreads();

    // ---- Phase B: softmax-max-normalize + adjacency mix + degree norm ----
    {
        const float lam = lam1[0];
        const float sg  = 1.f / (1.f + __expf(-lam));
        const float omg = 1.f - sg;
#pragma unroll
        for (int rr = 0; rr < 4; ++rr) {
            const int rl = wave * 4 + rr;         // local row 0..31
            const int ig = row0 + rl;             // global row (query index)
            const float qp0 = QP[((size_t)bh * SEQ + ig) * 2 + 0];
            const float qp1 = QP[((size_t)bh * SEQ + ig) * 2 + 1];
            float vals[16];
            float mx = -1e30f;
#pragma unroll
            for (int e = 0; e < 16; ++e) {
                vals[e] = sS[(size_t)rl * SPITCH + lane + 32 * e];
                mx = fmaxf(mx, vals[e]);
            }
#pragma unroll
            for (int off = 16; off >= 1; off >>= 1)
                mx = fmaxf(mx, __shfl_xor(mx, off, 32));
            float sum = 0.f;
#pragma unroll
            for (int e = 0; e < 16; ++e) {
                const int j = lane + 32 * e;
                const float p = sg * __expf(vals[e] - mx);   // attn / rowmax(attn)
                const float dj = (float)(ig - j);
                const float gco = (j < ig) ? qp1 : qp0;      // tril(-1)/triu(0)
                const float adj = __expf(gco * (-0.5f * dj * dj));
                const float a = p + omg * adj;
                vals[e] = a;
                sum += a;
            }
#pragma unroll
            for (int off = 16; off >= 1; off >>= 1)
                sum += __shfl_xor(sum, off, 32);
            const float inv = 1.f / (sum + 1e-9f);
#pragma unroll
            for (int e = 0; e < 16; ++e)
                sP[(size_t)rl * PPITCH + lane + 32 * e] = (half_t)(vals[e] * inv);
        }
    }
    __syncthreads();

    // ---- Phase C: O = A * V ------------------------------------------
    {
        const int rsub = wave & 1;          // 16-row half
        const int dsub = wave >> 1;         // 16-col group of DK=64
        v8f acc = {};
        for (int j0 = 0; j0 < SEQ; j0 += 32) {
            v16h a = load_afrag_h(sP + (size_t)(rsub * 16) * PPITCH, PPITCH, j0, lane);
            v16h b = load_bfrag_h(Vbh + (size_t)(dsub * 16) * SEQ, SEQ, j0, lane);
            acc = wmma16x16x32(a, b, acc);
        }
        const int mb = (lane >> 4) << 3;
        const int dloc = dsub * 16 + (lane & 15);
#pragma unroll
        for (int j = 0; j < 8; ++j) {
            const int tok = row0 + rsub * 16 + mb + j;
            Octx[((size_t)bb * SEQ + tok) * DMODEL + hh * DK + dloc] = (half_t)acc[j];
        }
    }
}

// ---------------------------------------------------------------------------
// Kernel 4: out = Octx @ Wo^T + bo  (f16 in, f32 out, same 32x64 wave tiling)
// ---------------------------------------------------------------------------
__global__ void __launch_bounds__(256)
out_gemm_kernel(const half_t* __restrict__ Xh, const half_t* __restrict__ W,
                const float* __restrict__ bias, float* __restrict__ out) {
    const int lane = threadIdx.x & 31;
    const int wave = threadIdx.x >> 5;
    const int m0 = blockIdx.y * 256 + wave * 32;
    const int n0 = blockIdx.x * 64;

    v8f acc[2][4];
#pragma unroll
    for (int r = 0; r < 2; ++r)
#pragma unroll
        for (int t = 0; t < 4; ++t) acc[r][t] = v8f{};

    const half_t* xrow0 = Xh + (size_t)m0 * DMODEL;
    const half_t* xrow1 = Xh + (size_t)(m0 + 16) * DMODEL;

    for (int k0 = 0; k0 < DMODEL; k0 += 32) {
        v16h a0 = load_afrag_h(xrow0, DMODEL, k0, lane);
        v16h a1 = load_afrag_h(xrow1, DMODEL, k0, lane);
#pragma unroll
        for (int t = 0; t < 4; ++t) {
            v16h b = load_bfrag_h(W + (size_t)(n0 + t * 16) * DMODEL, DMODEL, k0, lane);
            acc[0][t] = wmma16x16x32(a0, b, acc[0][t]);
            acc[1][t] = wmma16x16x32(a1, b, acc[1][t]);
        }
    }

    const int mb = (lane >> 4) << 3;
#pragma unroll
    for (int r = 0; r < 2; ++r) {
#pragma unroll
        for (int t = 0; t < 4; ++t) {
            const int n = n0 + t * 16 + (lane & 15);
            const float bvv = bias[n];
#pragma unroll
            for (int j = 0; j < 8; ++j) {
                const int m = m0 + r * 16 + mb + j;
                out[(size_t)m * DMODEL + n] = acc[r][t][j] + bvv;
            }
        }
    }
}

// ---------------------------------------------------------------------------
// Host-side launch
// ---------------------------------------------------------------------------
extern "C" void kernel_launch(void* const* d_in, const int* in_sizes, int n_in,
                              void* d_out, int out_size, void* d_ws, size_t ws_size,
                              hipStream_t stream) {
    (void)in_sizes; (void)n_in; (void)out_size; (void)ws_size;
    const float* query = (const float*)d_in[0];
    const float* key   = (const float*)d_in[1];
    const float* value = (const float*)d_in[2];
    const float* Wq = (const float*)d_in[3];
    const float* bq = (const float*)d_in[4];
    const float* Wk = (const float*)d_in[5];
    const float* bk = (const float*)d_in[6];
    const float* Wv = (const float*)d_in[7];
    const float* bv = (const float*)d_in[8];
    const float* Wg = (const float*)d_in[9];
    const float* bg = (const float*)d_in[10];
    const float* lam1 = (const float*)d_in[11];
    const float* Wo = (const float*)d_in[12];
    const float* bo = (const float*)d_in[13];
    float* out = (float*)d_out;

    // Workspace carve-up (all offsets 256B-aligned).
    char* ws = (char*)d_ws;
    const size_t ACT_BYTES = (size_t)NTOK * DMODEL * sizeof(half_t);   // 8 MB
    const size_t W_BYTES   = (size_t)DMODEL * DMODEL * sizeof(half_t); // 512 KB
    half_t* Xq   = (half_t*)(ws + 0 * ACT_BYTES);
    half_t* Xk   = (half_t*)(ws + 1 * ACT_BYTES);
    half_t* Xv   = (half_t*)(ws + 2 * ACT_BYTES);
    half_t* Qh   = (half_t*)(ws + 3 * ACT_BYTES);
    half_t* Kh   = (half_t*)(ws + 4 * ACT_BYTES);
    half_t* Vt   = (half_t*)(ws + 5 * ACT_BYTES);
    half_t* Wq16 = (half_t*)(ws + 6 * ACT_BYTES);
    half_t* Wk16 = (half_t*)(ws + 6 * ACT_BYTES + 1 * W_BYTES);
    half_t* Wv16 = (half_t*)(ws + 6 * ACT_BYTES + 2 * W_BYTES);
    half_t* Wo16 = (half_t*)(ws + 6 * ACT_BYTES + 3 * W_BYTES);
    float*  QP   = (float*)(ws + 6 * ACT_BYTES + 4 * W_BYTES);
    half_t* Octx = Xq;   // Xq is dead once Q-projection is done

    // f32 -> f16 conversions (one shot)
    const int actN8 = (NTOK * DMODEL) / 8;   // 524288
    const int wN8   = (DMODEL * DMODEL) / 8; // 32768
    cvt_f16_kernel<<<(actN8 + 255) / 256, 256, 0, stream>>>(query, Xq, actN8);
    cvt_f16_kernel<<<(actN8 + 255) / 256, 256, 0, stream>>>(key,   Xk, actN8);
    cvt_f16_kernel<<<(actN8 + 255) / 256, 256, 0, stream>>>(value, Xv, actN8);
    cvt_f16_kernel<<<(wN8 + 255) / 256, 256, 0, stream>>>(Wq, Wq16, wN8);
    cvt_f16_kernel<<<(wN8 + 255) / 256, 256, 0, stream>>>(Wk, Wk16, wN8);
    cvt_f16_kernel<<<(wN8 + 255) / 256, 256, 0, stream>>>(Wv, Wv16, wN8);
    cvt_f16_kernel<<<(wN8 + 255) / 256, 256, 0, stream>>>(Wo, Wo16, wN8);

    const dim3 gemm_grid(DMODEL / 64, NTOK / 256);   // 8 x 32
    proj_gemm_kernel<<<gemm_grid, 256, 0, stream>>>(Xq, Wq16, bq, Qh, 0);
    proj_gemm_kernel<<<gemm_grid, 256, 0, stream>>>(Xk, Wk16, bk, Kh, 0);
    proj_gemm_kernel<<<gemm_grid, 256, 0, stream>>>(Xv, Wv16, bv, Vt, 1);

    gamma_kernel<<<(BH * SEQ) / 256, 256, 0, stream>>>(Qh, Wg, bg, QP);

    attn_kernel<<<dim3(SEQ / 32, BH), 256, SMEM_TOTAL, stream>>>(
        Qh, Kh, Vt, QP, lam1, Octx);

    out_gemm_kernel<<<gemm_grid, 256, 0, stream>>>(Octx, Wo16, bo, out);
}
